// LSTM_87299505258616
// MI455X (gfx1250) — matrix operand order repaired and checked by reading, not compiled, not run on hardware
//
#include <hip/hip_runtime.h>

// ---------------------------------------------------------------------------
// Stacked bidirectional LSTM (2 stacks x 5 layers), B=256, T=1024, H=30.
// Latency-bound recurrence -> persistent per-layer kernel looping over T.
// gates^T = W_perm . [x;h]^T via v_wmma_f32_16x16x32_f16: A = permuted weights
// resident in VGPRs, B = x/h fragments from LDS, D = [gate x batch].
// Gate rows permuted (g' = 4j + gatetype) so each lane's 8 accumulator values
// are two complete units' (i,f,g,o) -> LSTM cell runs entirely in registers.
// ---------------------------------------------------------------------------

namespace {

constexpr int kB   = 256;   // batch
constexpr int kT   = 1024;  // time steps
constexpr int kIn0 = 21;    // stack-1 layer-0 input features
constexpr int kH   = 30;    // hidden units per direction
constexpr int kG   = 120;   // 4*H gate dim
constexpr int kL   = 5;     // layers per stack

constexpr int BM = 32;      // batch rows per block (2 N-tiles)
constexpr int NP = 128;     // padded gate dim (8 M-tiles of 16)
constexpr int KH = 32;      // padded hidden (1 K-step of 32)
constexpr int KX = 64;      // padded input features (2 K-steps of 32)

typedef _Float16 h16;
typedef __attribute__((ext_vector_type(16))) _Float16 v16h;
typedef __attribute__((ext_vector_type(8)))  _Float16 v8h;
typedef __attribute__((ext_vector_type(8)))  float    v8f;

// A-matrix (16x32, 16-bit) fragment gather from row-major LDS [rows][pitch].
// Lane L holds m = L%16; halves 0..7 -> k = kOff + sel*8 + h,
// halves 8..15 -> k = kOff + sel*8 + 16 + (h-8).  (ISA 7.12.2)
__device__ inline v16h load_fragA(const h16* base, int row, int pitch, int kOff) {
  const int lane = threadIdx.x & 31;
  const int r = lane & 15, sel = lane >> 4;
  const h16* p = base + (row + r) * pitch + kOff + sel * 8;
  v8h lo = *(const v8h*)(p);
  v8h hi = *(const v8h*)(p + 16);
  v16h f;
#pragma unroll
  for (int i = 0; i < 8; ++i) { f[i] = lo[i]; f[i + 8] = hi[i]; }
  return f;
}

// B-matrix (32x16, 16-bit) fragment gather. B[k][n] stored row-major as
// base[n][k]. Lane L holds n = L%16; halves 0..7 -> k = kOff + sel*8 + h,
// halves 8..15 -> k = kOff + sel*8 + 16 + (h-8).
__device__ inline v16h load_fragB(const h16* base, int row, int pitch, int kOff) {
  const int lane = threadIdx.x & 31;
  const int r = lane & 15, sel = lane >> 4;
  const h16* p = base + (row + r) * pitch + kOff + sel * 8;
  v8h lo = *(const v8h*)(p);
  v8h hi = *(const v8h*)(p + 16);
  v16h f;
#pragma unroll
  for (int i = 0; i < 8; ++i) { f[i] = lo[i]; f[i + 8] = hi[i]; }
  return f;
}

// Branch-free activations: v_exp_f32 + v_rcp_f32, clamped so exp stays finite.
__device__ inline float fast_sigmoid(float x) {
  x = fminf(fmaxf(x, -30.f), 30.f);
  return __builtin_amdgcn_rcpf(1.f + __expf(-x));
}
__device__ inline float fast_tanh(float x) {
  x = fminf(fmaxf(x, -15.f), 15.f);
  const float e = __expf(2.f * x);
  return (e - 1.f) * __builtin_amdgcn_rcpf(e + 1.f);
}

}  // namespace

// One bidirectional LSTM layer. grid = (kB/BM, 2 dirs), block = 256 (8 waves).
// blockIdx.y == 1 is the reverse direction (PyTorch dir index 1).
// Wave w owns gate M-tile w (permuted gate rows 16w..16w+15 = units 4w..4w+3).
__global__ __launch_bounds__(256) void lstm_layer_kernel(
    const float* __restrict__ xin, int F, int batchMajor,
    const float* __restrict__ WihBase,  // [2, G, F]
    const float* __restrict__ WhhBase,  // [2, G, H]
    const float* __restrict__ bBase,    // [2, G]
    const float* __restrict__ h0Base,   // [2, B, H]
    const float* __restrict__ c0Base,   // [2, B, H]
    float* __restrict__ xout)           // [T, B, 2H]
{
  __shared__ __align__(16) h16 sWih[NP * KX];  // 16 KB [gate'][k] permuted
  __shared__ __align__(16) h16 sWhh[NP * KH];  //  8 KB [gate'][k] permuted
  __shared__ __align__(16) h16 sX[BM * KX];    //  4 KB [batch][feature]
  __shared__ __align__(16) h16 sH[BM * KH];    //  2 KB [batch][unit]
  __shared__ float sBias[NP];                  // permuted bias

  const int tid = threadIdx.x;
  const int d   = blockIdx.y;           // direction
  const int bm  = blockIdx.x * BM;      // batch row base
  const int reverse = d;
  const int dirOff  = d * kH;

  const float* Wih  = WihBase + (size_t)d * kG * F;
  const float* Whh  = WhhBase + (size_t)d * kG * kH;
  const float* bias = bBase   + (size_t)d * kG;
  const float* h0   = h0Base  + (size_t)d * kB * kH;
  const float* c0   = c0Base  + (size_t)d * kB * kH;

  // ---- zero-fill LDS (covers all padding regions) ----
  for (int i = tid; i < NP * KX; i += 256) sWih[i] = (h16)0.f;
  for (int i = tid; i < NP * KH; i += 256) sWhh[i] = (h16)0.f;
  for (int i = tid; i < BM * KX; i += 256) sX[i]   = (h16)0.f;
  for (int i = tid; i < BM * KH; i += 256) sH[i]   = (h16)0.f;
  for (int i = tid; i < NP; i += 256)      sBias[i] = 0.f;
  __syncthreads();

  // ---- stage weights (gate-permuted, f16) / bias / h0 into LDS ----
  // source row n = gatetype*kH + j  ->  permuted row g' = 4*j + gatetype
  for (int i = tid; i < kG * F; i += 256) {
    int n = i / F, k = i - n * F;
    int g = n / kH, j = n - g * kH;
    sWih[(4 * j + g) * KX + k] = (h16)Wih[i];
  }
  for (int i = tid; i < kG * kH; i += 256) {
    int n = i / kH, k = i - n * kH;
    int g = n / kH, j = n - g * kH;
    sWhh[(4 * j + g) * KH + k] = (h16)Whh[i];
  }
  for (int i = tid; i < kG; i += 256) {
    int g = i / kH, j = i - g * kH;
    sBias[4 * j + g] = bias[i];
  }
  for (int i = tid; i < BM * kH; i += 256) {
    int m = i / kH, j = i - m * kH;
    sH[m * KH + j] = (h16)h0[(size_t)(bm + m) * kH + j];
  }
  __syncthreads();

  const int wave = tid >> 5;
  const int lane = tid & 31;
  const int r    = lane & 15;
  const int sel  = lane >> 4;

  // Resident A-fragments: this wave's 16 permuted gate rows.
  const v16h Aih0 = load_fragA(sWih, wave * 16, KX, 0);
  const v16h Aih1 = load_fragA(sWih, wave * 16, KX, 32);
  const v16h Ahh  = load_fragA(sWhh, wave * 16, KH, 0);

  // Per-lane gate biases for its 8 accumulator rows (row = 16w + 8*sel + v).
  float bgate[8];
#pragma unroll
  for (int v = 0; v < 8; ++v) bgate[v] = sBias[wave * 16 + sel * 8 + v];

  // Per-lane cell state: (nt, u) -> batch = nt*16 + r, unit = 4*wave + 2*sel + u.
  const int unitBase = 4 * wave + 2 * sel;
  float cst[4];
#pragma unroll
  for (int nt = 0; nt < 2; ++nt)
#pragma unroll
    for (int u = 0; u < 2; ++u) {
      const int unit = unitBase + u;
      cst[nt * 2 + u] =
          (unit < kH) ? c0[(size_t)(bm + nt * 16 + r) * kH + unit] : 0.f;
    }

  for (int s = 0; s < kT; ++s) {
    const int t = reverse ? (kT - 1 - s) : s;

    // ---- stage 1: x_t tile -> LDS (f16); power-of-two index math only ----
    if (batchMajor) {  // xin = [B, T, F] (raw model input), F = 21 <= 32
      const float* src = xin + (size_t)t * kIn0;
#pragma unroll
      for (int it = 0; it < (BM * 32) / 256; ++it) {
        const int p = tid + 256 * it;
        const int m = p >> 5, k = p & 31;
        if (k < kIn0)
          sX[m * KX + k] = (h16)src[(size_t)(bm + m) * (kT * kIn0) + k];
      }
    } else {           // xin = [T, B, 60] (previous layer output), F == 60
      const float* src = xin + ((size_t)t * kB + bm) * (2 * kH);
#pragma unroll
      for (int it = 0; it < (BM * KX) / 256; ++it) {
        const int p = tid + 256 * it;
        const int m = p >> 6, k = p & 63;
        if (k < 2 * kH)
          sX[m * KX + k] = (h16)src[m * (2 * kH) + k];
      }
    }
    __syncthreads();

    // ---- stage 2: gates^T tiles = W.[x;h]^T (WMMA, f32 accum, in-register) --
    v8f acc[2];
#pragma unroll
    for (int nt = 0; nt < 2; ++nt) {
      const v16h Bx0 = load_fragB(sX, nt * 16, KX, 0);
      const v16h Bx1 = load_fragB(sX, nt * 16, KX, 32);
      const v16h Bh  = load_fragB(sH, nt * 16, KH, 0);
      v8f a = {};
      a = __builtin_amdgcn_wmma_f32_16x16x32_f16(false, Aih0, false, Bx0,
                                                 (short)0, a, false, false);
      a = __builtin_amdgcn_wmma_f32_16x16x32_f16(false, Aih1, false, Bx1,
                                                 (short)0, a, false, false);
      a = __builtin_amdgcn_wmma_f32_16x16x32_f16(false, Ahh, false, Bh,
                                                 (short)0, a, false, false);
      acc[nt] = a;
    }
    // All waves must finish reading sX/sH fragments before sH is rewritten.
    __syncthreads();

    // ---- stage 3: LSTM cell fully in registers on the accumulators ----
#pragma unroll
    for (int nt = 0; nt < 2; ++nt) {
#pragma unroll
      for (int u = 0; u < 2; ++u) {
        const float gi = acc[nt][4 * u + 0] + bgate[4 * u + 0];
        const float gf = acc[nt][4 * u + 1] + bgate[4 * u + 1];
        const float gg = acc[nt][4 * u + 2] + bgate[4 * u + 2];
        const float go = acc[nt][4 * u + 3] + bgate[4 * u + 3];
        const float c =
            fast_sigmoid(gf) * cst[nt * 2 + u] + fast_sigmoid(gi) * fast_tanh(gg);
        const float h = fast_sigmoid(go) * fast_tanh(c);
        cst[nt * 2 + u] = c;
        const int unit = unitBase + u;
        if (unit < kH) {
          const int batch = nt * 16 + r;
          sH[batch * KH + unit] = (h16)h;
          xout[((size_t)t * kB + (bm + batch)) * (2 * kH) + dirOff + unit] = h;
        }
      }
    }
    // Next iteration's post-stage-1 barrier orders these sH writes before the
    // next step's fragment loads.
  }
}

// Final FC: out[b] = tanh(xs[T-1, b, :] @ fc_w^T + fc_b), fc_w = [2, 60].
__global__ __launch_bounds__(256) void fc_kernel(
    const float* __restrict__ last,   // [B, 60]
    const float* __restrict__ fc_w,   // [2, 60]
    const float* __restrict__ fc_b,   // [2]
    float* __restrict__ out)          // [B, 2]
{
  const int b = blockIdx.x * blockDim.x + threadIdx.x;
  if (b < kB) {
    const float* row = last + (size_t)b * (2 * kH);
    float a0 = fc_b[0], a1 = fc_b[1];
#pragma unroll
    for (int k = 0; k < 2 * kH; ++k) {
      const float v = row[k];
      a0 += v * fc_w[k];
      a1 += v * fc_w[2 * kH + k];
    }
    out[b * 2 + 0] = fast_tanh(a0);
    out[b * 2 + 1] = fast_tanh(a1);
  }
}

extern "C" void kernel_launch(void* const* d_in, const int* in_sizes, int n_in,
                              void* d_out, int out_size, void* d_ws, size_t ws_size,
                              hipStream_t stream) {
  (void)in_sizes; (void)n_in; (void)out_size; (void)ws_size;

  const float* x           = (const float*)d_in[0];   // [B, T, 21]
  const float* w_ih1_l0    = (const float*)d_in[1];   // [2, G, 21]
  const float* w_ih1_rest  = (const float*)d_in[2];   // [L-1, 2, G, 60]
  const float* w_hh1       = (const float*)d_in[3];   // [L, 2, G, H]
  const float* b1          = (const float*)d_in[4];   // [L, 2, G]
  const float* w_ih2       = (const float*)d_in[5];   // [L, 2, G, 60]
  const float* w_hh2       = (const float*)d_in[6];   // [L, 2, G, H]
  const float* b2          = (const float*)d_in[7];   // [L, 2, G]
  const float* h01         = (const float*)d_in[8];   // [2L, B, H]
  const float* c01         = (const float*)d_in[9];
  const float* h02         = (const float*)d_in[10];
  const float* c02         = (const float*)d_in[11];
  const float* fc_w        = (const float*)d_in[12];  // [2, 60]
  const float* fc_b        = (const float*)d_in[13];  // [2]
  float* out               = (float*)d_out;

  const size_t seqElems = (size_t)kT * kB * (2 * kH);  // 15.7M floats
  float* buf0 = (float*)d_ws;
  float* buf1 = buf0 + seqElems;

  const dim3 grid(kB / BM, 2);
  const dim3 blk(256);

  const float* cur = x;
  float* nxt = buf0;

  // ---- stack 1 ----
  for (int l = 0; l < kL; ++l) {
    const float* WihB;
    int F, bMajor;
    if (l == 0) { WihB = w_ih1_l0; F = kIn0; bMajor = 1; }
    else        { WihB = w_ih1_rest + (size_t)(l - 1) * 2 * kG * (2 * kH); F = 2 * kH; bMajor = 0; }
    lstm_layer_kernel<<<grid, blk, 0, stream>>>(
        cur, F, bMajor, WihB,
        w_hh1 + (size_t)l * 2 * kG * kH,
        b1    + (size_t)l * 2 * kG,
        h01   + (size_t)l * 2 * kB * kH,
        c01   + (size_t)l * 2 * kB * kH,
        nxt);
    cur = nxt;
    nxt = (nxt == buf0) ? buf1 : buf0;
  }

  // ---- stack 2 ----
  for (int l = 0; l < kL; ++l) {
    lstm_layer_kernel<<<grid, blk, 0, stream>>>(
        cur, 2 * kH, 0,
        w_ih2 + (size_t)l * 2 * kG * (2 * kH),
        w_hh2 + (size_t)l * 2 * kG * kH,
        b2    + (size_t)l * 2 * kG,
        h02   + (size_t)l * 2 * kB * kH,
        c02   + (size_t)l * 2 * kB * kH,
        nxt);
    cur = nxt;
    nxt = (nxt == buf0) ? buf1 : buf0;
  }

  // ---- final FC on xs[T-1] ----
  fc_kernel<<<dim3(1), blk, 0, stream>>>(
      cur + (size_t)(kT - 1) * kB * (2 * kH), fc_w, fc_b, out);
}
